// CausalSelfAttentionALiBi_47038481825982
// MI455X (gfx1250) — matrix operand order repaired and checked by reading, not compiled, not run on hardware
//
#include <hip/hip_runtime.h>

// ---------------------------------------------------------------------------
// CDNA5 / gfx1250 fused attention (ALiBi + sliding-window causal), wave32 WMMA
// ---------------------------------------------------------------------------

typedef __attribute__((ext_vector_type(16))) _Float16 v16h;
typedef __attribute__((ext_vector_type(8)))  _Float16 v8h;
typedef __attribute__((ext_vector_type(4)))  _Float16 v4h;
typedef __attribute__((ext_vector_type(8)))  float    v8f;

#define B_    4
#define L_    2048
#define D_    1024
#define H_    16
#define HD_   64
#define WIN_  512

static __device__ __forceinline__ v8f wmma_f16(v16h a, v16h b, v8f c) {
  // D = A(16x32 f16) * B(32x16 f16) + C(16x16 f32)
  return __builtin_amdgcn_wmma_f32_16x16x32_f16(
      /*neg_a=*/false, a, /*neg_b=*/false, b,
      /*c_mod=*/(short)0, c, /*reuse_a=*/false, /*reuse_b=*/false);
}

union ABfrag { v16h v; v8h h[2]; };

// DPP ROW_XMASK lane-xor within 16-lane rows (pure VALU, no LDS traffic).
template <int XM>
static __device__ __forceinline__ float dpp_xor(float v) {
  int r = __builtin_amdgcn_update_dpp(0, __float_as_int(v),
                                      0x160 | XM, 0xf, 0xf, true);
  return __int_as_float(r);
}
static __device__ __forceinline__ float row_max16(float v) {
  v = fmaxf(v, dpp_xor<1>(v));
  v = fmaxf(v, dpp_xor<2>(v));
  v = fmaxf(v, dpp_xor<4>(v));
  v = fmaxf(v, dpp_xor<8>(v));
  return v;
}
static __device__ __forceinline__ float row_sum16(float v) {
  v += dpp_xor<1>(v);
  v += dpp_xor<2>(v);
  v += dpp_xor<4>(v);
  v += dpp_xor<8>(v);
  return v;
}

// ---------------------------------------------------------------------------
// f32 -> f16 conversion (vectorized x4)
// ---------------------------------------------------------------------------
__global__ void cvt4_kernel(const float* __restrict__ src,
                            _Float16* __restrict__ dst, int n4) {
  int i = blockIdx.x * blockDim.x + threadIdx.x;
  if (i < n4) {
    float4 v = ((const float4*)src)[i];
    v4h h;
    h[0] = (_Float16)v.x; h[1] = (_Float16)v.y;
    h[2] = (_Float16)v.z; h[3] = (_Float16)v.w;
    ((v4h*)dst)[i] = h;
  }
}

// ---------------------------------------------------------------------------
// GEMM: Y[m,n] = sum_k A[m,k] * W[n,k] + bias[n]
// Register-blocked: one wave computes a 32x64 output block (2 M-tiles x
// 4 N-tiles = 8 WMMA accumulators), k-steps of 32.
// MODE 0: store Q*0.125 -> f16 [B,H,L,d]
// MODE 1: store K       -> f16 [B,H,L,d]
// MODE 2: store V^T     -> f16 [B,H,d,L]
// MODE 3: store O       -> f32 [B*L, D]
// ---------------------------------------------------------------------------
template <int MODE>
__global__ __launch_bounds__(256) void gemm_kernel(
    const _Float16* __restrict__ A, const _Float16* __restrict__ W,
    const float* __restrict__ bias, _Float16* __restrict__ oh,
    float* __restrict__ of) {
  const int lane = threadIdx.x & 31;
  const int wv   = threadIdx.x >> 5;
  const int col  = lane & 15;
  const int hi   = lane >> 4;
  const int n0   = (blockIdx.x * 8 + wv) * 64;  // 64-wide N block
  const int m0   = blockIdx.y * 32;             // 32-tall M block

  // A fragments: rows m0+col (tile 0) and m0+16+col (tile 1)
  const _Float16* ap0 = A + (size_t)(m0 + col) * D_ + hi * 16;
  const _Float16* ap1 = ap0 + (size_t)16 * D_;
  // B fragments: columns n0 + 16*j + col, contiguous K run of 16
  const _Float16* wp = W + (size_t)(n0 + col) * D_ + hi * 16;

  v8f acc[2][4] = {};
#pragma unroll 2
  for (int kc = 0; kc < D_; kc += 32) {
    ABfrag a0, a1;
    a0.h[0] = *(const v8h*)(ap0 + kc);
    a0.h[1] = *(const v8h*)(ap0 + kc + 8);
    a1.h[0] = *(const v8h*)(ap1 + kc);
    a1.h[1] = *(const v8h*)(ap1 + kc + 8);
    v16h b0 = *(const v16h*)(wp + kc);
    v16h b1 = *(const v16h*)(wp + (size_t)16 * D_ + kc);
    v16h b2 = *(const v16h*)(wp + (size_t)32 * D_ + kc);
    v16h b3 = *(const v16h*)(wp + (size_t)48 * D_ + kc);
    acc[0][0] = wmma_f16(a0.v, b0, acc[0][0]);
    acc[0][1] = wmma_f16(a0.v, b1, acc[0][1]);
    acc[0][2] = wmma_f16(a0.v, b2, acc[0][2]);
    acc[0][3] = wmma_f16(a0.v, b3, acc[0][3]);
    acc[1][0] = wmma_f16(a1.v, b0, acc[1][0]);
    acc[1][1] = wmma_f16(a1.v, b1, acc[1][1]);
    acc[1][2] = wmma_f16(a1.v, b2, acc[1][2]);
    acc[1][3] = wmma_f16(a1.v, b3, acc[1][3]);
  }

#pragma unroll
  for (int j = 0; j < 4; ++j) {
    const int n = n0 + j * 16 + col;
    const float bn = bias[n];
#pragma unroll
    for (int i = 0; i < 2; ++i) {
#pragma unroll
      for (int r = 0; r < 8; ++r) {
        const int tok = m0 + i * 16 + r + hi * 8;  // global token (b*L + i)
        const float val = acc[i][j][r] + bn;
        if (MODE == 3) {
          of[(size_t)tok * D_ + n] = val;
        } else {
          const int bb = tok >> 11;  // / L_
          const int ii = tok & (L_ - 1);
          const int hh = n >> 6;
          const int dv = n & 63;
          if (MODE == 2) {
            // V^T: [B,H,d,L]
            oh[((size_t)(bb * H_ + hh) * HD_ + dv) * L_ + ii] = (_Float16)val;
          } else {
            // Q/K: [B,H,L,d]
            const float s = (MODE == 0) ? val * 0.125f : val;  // 1/sqrt(64)
            oh[(((size_t)(bb * H_ + hh) * L_ + ii) << 6) + dv] = (_Float16)s;
          }
        }
      }
    }
  }
}

// ---------------------------------------------------------------------------
// Flash-style windowed attention with ALiBi. One wave per (b, h, 16-query
// tile); 32-key blocks; online softmax with DPP row reductions; P transposed
// via per-wave LDS (C-layout -> A-layout).
// ---------------------------------------------------------------------------
__global__ __launch_bounds__(256) void attn_kernel(
    const _Float16* __restrict__ Qh, const _Float16* __restrict__ Kh,
    const _Float16* __restrict__ Vt, const unsigned char* __restrict__ pad,
    _Float16* __restrict__ Ah) {
  __shared__ _Float16 Pr[8][16][32];  // per-wave P tile (row-major), 8 KB

  const int lane = threadIdx.x & 31;
  const int wv   = threadIdx.x >> 5;
  const int col  = lane & 15;
  const int hi   = lane >> 4;

  const int gw = blockIdx.x * 8 + wv;      // 0..8191
  const int b  = gw >> 11;                 // / 2048
  const int h  = (gw >> 7) & 15;
  const int i0 = (gw & 127) << 4;
  const int bh = b * H_ + h;
  const float slope = exp2f(-0.5f * (float)(h + 1));  // ALiBi, H=16

  // Q A-fragments for K-dim chunks [0,32) and [32,64)
  const _Float16* qp = Qh + ((size_t)(bh * L_ + i0 + col)) * HD_ + hi * 16;
  ABfrag aq0, aq1;
  aq0.h[0] = *(const v8h*)(qp);      aq0.h[1] = *(const v8h*)(qp + 8);
  aq1.h[0] = *(const v8h*)(qp + 32); aq1.h[1] = *(const v8h*)(qp + 40);

  float mrow[8], lrow[8];
  v8f o0 = {}, o1 = {}, o2 = {}, o3 = {};
#pragma unroll
  for (int r = 0; r < 8; ++r) { mrow[r] = -1e30f; lrow[r] = 0.0f; }

  int jlo = i0 - WIN_; if (jlo < 0) jlo = 0;
  const int jb0  = jlo & ~31;
  const int nblk = (i0 + 16 - jb0 + 31) >> 5;

  const _Float16* kbh = Kh + (size_t)bh * L_ * HD_;
  const _Float16* vbh = Vt + (size_t)bh * HD_ * L_;
  const unsigned char* padb = pad + b * L_;

  for (int t = 0; t < nblk; ++t) {
    const int j0 = jb0 + t * 32;

    // --- K B-fragments (key rows, contiguous along d) ---
    int kr0 = j0 + col;      if (kr0 > L_ - 1) kr0 = L_ - 1;
    int kr1 = j0 + 16 + col; if (kr1 > L_ - 1) kr1 = L_ - 1;
    const _Float16* kp0 = kbh + (size_t)kr0 * HD_ + hi * 16;
    const _Float16* kp1 = kbh + (size_t)kr1 * HD_ + hi * 16;
    v16h bk00 = *(const v16h*)(kp0);
    v16h bk01 = *(const v16h*)(kp0 + 32);
    v16h bk10 = *(const v16h*)(kp1);
    v16h bk11 = *(const v16h*)(kp1 + 32);

    // --- S = (Q/sqrt(d)) K^T : two 16x16 f32 tiles ---
    v8f s0 = {}, s1 = {};
    s0 = wmma_f16(aq0.v, bk00, s0);
    s0 = wmma_f16(aq1.v, bk01, s0);
    s1 = wmma_f16(aq0.v, bk10, s1);
    s1 = wmma_f16(aq1.v, bk11, s1);

    const int ja = j0 + col, jb = j0 + 16 + col;
    const bool pma = padb[kr0] != 0;
    const bool pmb = padb[kr1] != 0;

    float sv0[8], sv1[8];
#pragma unroll
    for (int r = 0; r < 8; ++r) {
      const int irow = i0 + r + hi * 8;
      float a = s0[r] + slope * (float)(ja - irow);
      float c = s1[r] + slope * (float)(jb - irow);
      if (ja > irow || ja < irow - WIN_ || pma) a = -__builtin_inff();
      if (jb > irow || jb < irow - WIN_ || pmb) c = -__builtin_inff();
      sv0[r] = a; sv1[r] = c;
    }

    // --- online softmax (DPP ROW_XMASK reductions inside 16-lane rows) ---
    float fct[8];
#pragma unroll
    for (int r = 0; r < 8; ++r) {
      const float vm = row_max16(fmaxf(sv0[r], sv1[r]));
      const float mn = fmaxf(mrow[r], vm);
      const float f  = __expf(mrow[r] - mn);
      const float e0 = __expf(sv0[r] - mn);
      const float e1 = __expf(sv1[r] - mn);
      const float rs = row_sum16(e0 + e1);
      lrow[r] = lrow[r] * f + rs;
      mrow[r] = mn;
      fct[r]  = f;
      // store P tile row-major into this wave's LDS slab (C->A transpose)
      const int row = r + hi * 8;
      Pr[wv][row][col]      = (_Float16)e0;
      Pr[wv][row][col + 16] = (_Float16)e1;
    }
#pragma unroll
    for (int r = 0; r < 8; ++r) {
      o0[r] *= fct[r]; o1[r] *= fct[r]; o2[r] *= fct[r]; o3[r] *= fct[r];
    }

    __builtin_amdgcn_wave_barrier();  // keep ds_store before ds_load (in-order LDS)
    ABfrag pa;
    pa.h[0] = *(const v8h*)&Pr[wv][col][hi * 16];
    pa.h[1] = *(const v8h*)&Pr[wv][col][hi * 16 + 8];
    __builtin_amdgcn_wave_barrier();  // loads complete before next-iter stores

    // --- V^T B-fragments: column dv, contiguous along keys ---
    int jv = j0 + hi * 16; if (jv > L_ - 16) jv = L_ - 16;  // masked keys: P==0
    const _Float16* vp = vbh + (size_t)col * L_ + jv;
    v16h bv0 = *(const v16h*)(vp);
    v16h bv1 = *(const v16h*)(vp + 16 * L_);
    v16h bv2 = *(const v16h*)(vp + 32 * L_);
    v16h bv3 = *(const v16h*)(vp + 48 * L_);

    o0 = wmma_f16(pa.v, bv0, o0);
    o1 = wmma_f16(pa.v, bv1, o1);
    o2 = wmma_f16(pa.v, bv2, o2);
    o3 = wmma_f16(pa.v, bv3, o3);
  }

  // --- normalize & store attention output as f16 [B, L, D] (head-concat) ---
#pragma unroll
  for (int r = 0; r < 8; ++r) {
    const float inv = 1.0f / fmaxf(lrow[r], 1e-20f);
    const int row = r + hi * 8;
    _Float16* op = Ah + ((size_t)(b * L_ + i0 + row)) * D_ + h * HD_ + col;
    op[0]  = (_Float16)(o0[r] * inv);
    op[16] = (_Float16)(o1[r] * inv);
    op[32] = (_Float16)(o2[r] * inv);
    op[48] = (_Float16)(o3[r] * inv);
  }
}

// ---------------------------------------------------------------------------
// Launch
// ---------------------------------------------------------------------------
extern "C" void kernel_launch(void* const* d_in, const int* in_sizes, int n_in,
                              void* d_out, int out_size, void* d_ws,
                              size_t ws_size, hipStream_t stream) {
  const float* x          = (const float*)d_in[0];
  const unsigned char* km = (const unsigned char*)d_in[1];
  const float* Wq = (const float*)d_in[2];
  const float* bq = (const float*)d_in[3];
  const float* Wk = (const float*)d_in[4];
  const float* bk = (const float*)d_in[5];
  const float* Wv = (const float*)d_in[6];
  const float* bv = (const float*)d_in[7];
  const float* Wo = (const float*)d_in[8];
  const float* bo = (const float*)d_in[9];
  float* out = (float*)d_out;

  const size_t NXD = (size_t)B_ * L_ * D_;  // 8388608
  const size_t NW  = (size_t)D_ * D_;       // 1048576

  _Float16* p   = (_Float16*)d_ws;
  _Float16* Xh  = p; p += NXD;
  _Float16* Wqh = p; p += NW;
  _Float16* Wkh = p; p += NW;
  _Float16* Wvh = p; p += NW;
  _Float16* Woh = p; p += NW;
  _Float16* Qh  = p; p += NXD;
  _Float16* Kh  = p; p += NXD;
  _Float16* Vt  = p; p += NXD;
  _Float16* Ah  = p; p += NXD;

  // f32 -> f16 conversions
  cvt4_kernel<<<dim3((unsigned)(NXD / 4 / 256)), 256, 0, stream>>>(x, Xh, (int)(NXD / 4));
  cvt4_kernel<<<dim3((unsigned)(NW / 4 / 256)), 256, 0, stream>>>(Wq, Wqh, (int)(NW / 4));
  cvt4_kernel<<<dim3((unsigned)(NW / 4 / 256)), 256, 0, stream>>>(Wk, Wkh, (int)(NW / 4));
  cvt4_kernel<<<dim3((unsigned)(NW / 4 / 256)), 256, 0, stream>>>(Wv, Wvh, (int)(NW / 4));
  cvt4_kernel<<<dim3((unsigned)(NW / 4 / 256)), 256, 0, stream>>>(Wo, Woh, (int)(NW / 4));

  // Projections (WMMA, 32x64 per wave): grid = (N/64/8, M/32)
  dim3 ggrid(2, (B_ * L_) / 32, 1);
  gemm_kernel<0><<<ggrid, 256, 0, stream>>>(Xh, Wqh, bq, Qh, nullptr);
  gemm_kernel<1><<<ggrid, 256, 0, stream>>>(Xh, Wkh, bk, Kh, nullptr);
  gemm_kernel<2><<<ggrid, 256, 0, stream>>>(Xh, Wvh, bv, Vt, nullptr);

  // Windowed ALiBi attention: 8192 waves (one per 16-query tile)
  attn_kernel<<<dim3(1024), 256, 0, stream>>>(Qh, Kh, Vt, km, Ah);

  // Output projection -> f32
  gemm_kernel<3><<<ggrid, 256, 0, stream>>>(Ah, Woh, bo, nullptr, out);
}
